// Relational_44736379355236
// MI455X (gfx1250) — matrix operand order repaired and checked by reading, not compiled, not run on hardware
//
#include <hip/hip_runtime.h>
#include <hip/hip_bf16.h>

typedef __attribute__((ext_vector_type(16))) __bf16 v16bf;
typedef __attribute__((ext_vector_type(8)))  float  v8f;
typedef __attribute__((ext_vector_type(4)))  int    v4i;

#define HEADS 16
#define DIM   1024
#define NE    2048
#define BATCH 8
#define ROWS  (BATCH * NE)      // 16384
#define TM 128
#define TN 128
#define TK 32
#define KSTEPS (DIM / TK)       // 32
#define LDSB 80                 // padded row stride in bytes (40 bf16) -> conflict-free b128 reads

#if defined(__AMDGCN__) && __has_builtin(__builtin_amdgcn_global_load_async_to_lds_b128)
#define HAVE_ASYNC 1
#else
#define HAVE_ASYNC 0
#endif

#if HAVE_ASYNC
typedef __attribute__((address_space(1))) v4i* gv4i_ptr;
typedef __attribute__((address_space(3))) v4i* lv4i_ptr;
__device__ static inline gv4i_ptr as_global(const void* p) {
    return (gv4i_ptr)(void*)p;
}
__device__ static inline lv4i_ptr as_lds(void* p) {
    return (lv4i_ptr)p;
}
__device__ static inline void wait_async0() {
#if __has_builtin(__builtin_amdgcn_s_wait_asynccnt)
    __builtin_amdgcn_s_wait_asynccnt(0);
#else
    asm volatile("s_wait_asynccnt 0x0" ::: "memory");
#endif
}
#endif

// ---------------- Kernel 1: LayerNorm + convert to bf16 ----------------
__global__ __launch_bounds__(256) void ln_bf16_kernel(
    const float* __restrict__ inp, const float* __restrict__ gamma,
    const float* __restrict__ beta, __hip_bfloat16* __restrict__ xout)
{
    const int row = blockIdx.x;
    const int t   = threadIdx.x;
    const float4 x = ((const float4*)(inp + (size_t)row * DIM))[t];

    float s  = x.x + x.y + x.z + x.w;
    float s2 = x.x*x.x + x.y*x.y + x.z*x.z + x.w*x.w;
    #pragma unroll
    for (int off = 16; off > 0; off >>= 1) {
        s  += __shfl_xor(s,  off, 32);
        s2 += __shfl_xor(s2, off, 32);
    }
    __shared__ float ws[8], ws2[8];
    const int lane = t & 31, wv = t >> 5;
    if (lane == 0) { ws[wv] = s; ws2[wv] = s2; }
    __syncthreads();
    if (t == 0) {
        float a = 0.f, b = 0.f;
        #pragma unroll
        for (int i = 0; i < 8; ++i) { a += ws[i]; b += ws2[i]; }
        ws[0] = a; ws2[0] = b;
    }
    __syncthreads();
    const float mean = ws[0]  * (1.0f / DIM);
    const float var  = ws2[0] * (1.0f / DIM) - mean * mean;
    const float rstd = rsqrtf(var + 1e-5f);

    const float4 g = ((const float4*)gamma)[t];
    const float4 b = ((const float4*)beta)[t];
    union { __hip_bfloat16 h[4]; uint2 u; } p;
    p.h[0] = __float2bfloat16((x.x - mean) * rstd * g.x + b.x);
    p.h[1] = __float2bfloat16((x.y - mean) * rstd * g.y + b.y);
    p.h[2] = __float2bfloat16((x.z - mean) * rstd * g.z + b.z);
    p.h[3] = __float2bfloat16((x.w - mean) * rstd * g.w + b.w);
    ((uint2*)(xout + (size_t)row * DIM))[t] = p.u;
}

// ---------------- Kernel 2: transpose W[K,N] -> Wt[N,K] in bf16 ----------------
__global__ __launch_bounds__(256) void wt_bf16_kernel(
    const float* __restrict__ W, __hip_bfloat16* __restrict__ Wt, int K, int N)
{
    __shared__ float tile[32][33];
    const int n0 = blockIdx.x * 32, k0 = blockIdx.y * 32;
    const int tx = threadIdx.x & 31, ty = threadIdx.x >> 5;   // 32 x 8
    #pragma unroll
    for (int i = 0; i < 4; ++i) {
        const int k = k0 + ty + i * 8;
        tile[ty + i * 8][tx] = W[(size_t)k * N + n0 + tx];
    }
    __syncthreads();
    #pragma unroll
    for (int i = 0; i < 4; ++i) {
        const int n = n0 + ty + i * 8;
        Wt[(size_t)n * K + k0 + tx] = __float2bfloat16(tile[tx][ty + i * 8]);
    }
}

// ---------------- Kernel 3: WMMA GEMM + qkv scatter epilogue ----------------
// X: [ROWS, DIM] bf16 row-major. Wt: [N, DIM] bf16 row-major (pre-transposed).
// mode 0: N=2048 fused qk projection (interleaved q/k).  mode 1: N=1024 v projection.
__global__ __launch_bounds__(256) void gemm_qkv_kernel(
    const __hip_bfloat16* __restrict__ X,
    const __hip_bfloat16* __restrict__ Wt,
    const float* __restrict__ bias,
    float* __restrict__ out_q, float* __restrict__ out_k, float* __restrict__ out_v,
    int mode)
{
    __shared__ __align__(16) char ldsA[2][TM * LDSB];
    __shared__ __align__(16) char ldsB[2][TN * LDSB];

    const int t     = threadIdx.x;
    const int m0    = blockIdx.y * TM;
    const int n0    = blockIdx.x * TN;
    const int lane  = t & 31, wave = t >> 5;
    const int waveM = wave & 3, waveN = wave >> 2;     // waves: 4 (M) x 2 (N)
    const int hlf   = lane >> 4, lnn = lane & 15;

    // tile loader mapping: 512 16B chunks per matrix, 2 per thread
    const int lrow = t >> 2;   // 0..63 (and +64)
    const int lseg = t & 3;    // 16B segment within a 64B k-row

    const __hip_bfloat16* gA0 = X  + (size_t)(m0 + lrow)      * DIM + lseg * 8;
    const __hip_bfloat16* gA1 = X  + (size_t)(m0 + lrow + 64) * DIM + lseg * 8;
    const __hip_bfloat16* gB0 = Wt + (size_t)(n0 + lrow)      * DIM + lseg * 8;
    const __hip_bfloat16* gB1 = Wt + (size_t)(n0 + lrow + 64) * DIM + lseg * 8;
    const int lOffA0 = lrow * LDSB + lseg * 16;
    const int lOffA1 = (lrow + 64) * LDSB + lseg * 16;

    v8f acc[2][4];
    #pragma unroll
    for (int mt = 0; mt < 2; ++mt)
        #pragma unroll
        for (int nt = 0; nt < 4; ++nt)
            acc[mt][nt] = (v8f){0.f,0.f,0.f,0.f,0.f,0.f,0.f,0.f};

    union FragBF { uint4 u[2]; v16bf v; };

#if HAVE_ASYNC
    // --- CDNA5 async global->LDS copy path (ASYNCcnt) ---
    auto asyncTile = [&](int kk, int buf) {
        __builtin_amdgcn_global_load_async_to_lds_b128(as_global(gA0 + kk), as_lds(ldsA[buf] + lOffA0), 0, 0);
        __builtin_amdgcn_global_load_async_to_lds_b128(as_global(gA1 + kk), as_lds(ldsA[buf] + lOffA1), 0, 0);
        __builtin_amdgcn_global_load_async_to_lds_b128(as_global(gB0 + kk), as_lds(ldsB[buf] + lOffA0), 0, 0);
        __builtin_amdgcn_global_load_async_to_lds_b128(as_global(gB1 + kk), as_lds(ldsB[buf] + lOffA1), 0, 0);
    };
    asyncTile(0, 0);
#else
    // --- fallback: reg-staged global->LDS copy ---
    uint4 ra0, ra1, rb0, rb1;
    auto loadG = [&](int kk) {
        ra0 = *(const uint4*)(gA0 + kk);
        ra1 = *(const uint4*)(gA1 + kk);
        rb0 = *(const uint4*)(gB0 + kk);
        rb1 = *(const uint4*)(gB1 + kk);
    };
    auto storeL = [&](int buf) {
        *(uint4*)(ldsA[buf] + lOffA0) = ra0;
        *(uint4*)(ldsA[buf] + lOffA1) = ra1;
        *(uint4*)(ldsB[buf] + lOffA0) = rb0;
        *(uint4*)(ldsB[buf] + lOffA1) = rb1;
    };
    loadG(0);
    storeL(0);
#endif

    for (int kt = 0; kt < KSTEPS; ++kt) {
        const int cur = kt & 1;
#if HAVE_ASYNC
        wait_async0();            // this wave's copies for tile kt have landed in LDS
        __syncthreads();          // everyone's copies landed; everyone done reading buf[1-cur]
        if (kt + 1 < KSTEPS) asyncTile((kt + 1) * TK, 1 - cur);  // overlaps with compute below
#else
        if (kt + 1 < KSTEPS) loadG((kt + 1) * TK);
        __syncthreads();
#endif

        const char* pa = ldsA[cur];
        const char* pb = ldsB[cur];
        FragBF a[2], b[4];
        #pragma unroll
        for (int mt = 0; mt < 2; ++mt) {
            const int r = waveM * 32 + mt * 16 + lnn;
            // A frag (16x32 bf16): lane row M=lnn, K = {hlf*8+0..7, 16+hlf*8+0..7}
            a[mt].u[0] = *(const uint4*)(pa + r * LDSB + hlf * 16);
            a[mt].u[1] = *(const uint4*)(pa + r * LDSB + 32 + hlf * 16);
        }
        #pragma unroll
        for (int nt = 0; nt < 4; ++nt) {
            const int n = waveN * 64 + nt * 16 + lnn;
            // B frag (32x16 bf16): lane col N=lnn, K = {hlf*8+0..7, 16+hlf*8+0..7} (K-major Wt)
            b[nt].u[0] = *(const uint4*)(pb + n * LDSB + hlf * 16);
            b[nt].u[1] = *(const uint4*)(pb + n * LDSB + 32 + hlf * 16);
        }
        #pragma unroll
        for (int mt = 0; mt < 2; ++mt)
            #pragma unroll
            for (int nt = 0; nt < 4; ++nt)
                acc[mt][nt] = __builtin_amdgcn_wmma_f32_16x16x32_bf16(
                    false, a[mt].v, false, b[nt].v, (short)0, acc[mt][nt], false, false);

#if !HAVE_ASYNC
        if (kt + 1 < KSTEPS) storeL(1 - cur);
#endif
    }

    // Epilogue: bias + scatter.  D-matrix layout: element i -> M = hlf*8 + i, N = lnn.
    #pragma unroll
    for (int mt = 0; mt < 2; ++mt) {
        const int rbase = m0 + waveM * 32 + mt * 16 + hlf * 8;
        #pragma unroll
        for (int nt = 0; nt < 4; ++nt) {
            const int e  = n0 + waveN * 64 + nt * 16 + lnn;
            const float bv = bias[e];
            if (mode == 0) {
                const int h = e >> 7, j = (e >> 1) & 63;
                const bool isK = (e & 1) != 0;
                #pragma unroll
                for (int i = 0; i < 8; ++i) {
                    const int r  = rbase + i;
                    const int bb = r >> 11, n = r & (NE - 1);
                    const float val = acc[mt][nt][i] + bv;
                    if (isK)
                        out_k[(((size_t)bb * HEADS + h) * 64 + j) * NE + n] = val;
                    else
                        out_q[(((size_t)bb * HEADS + h) * NE + n) * 64 + j] = val;
                }
            } else {
                const int h = e >> 6, j = e & 63;
                #pragma unroll
                for (int i = 0; i < 8; ++i) {
                    const int r  = rbase + i;
                    const int bb = r >> 11, n = r & (NE - 1);
                    out_v[(((size_t)bb * HEADS + h) * NE + n) * 64 + j] = acc[mt][nt][i] + bv;
                }
            }
        }
    }
}

extern "C" void kernel_launch(void* const* d_in, const int* in_sizes, int n_in,
                              void* d_out, int out_size, void* d_ws, size_t ws_size,
                              hipStream_t stream)
{
    const float* inp   = (const float*)d_in[0];
    const float* gamma = (const float*)d_in[1];
    const float* beta  = (const float*)d_in[2];
    const float* W_qk  = (const float*)d_in[3];
    const float* b_qk  = (const float*)d_in[4];
    const float* W_v   = (const float*)d_in[5];
    const float* b_v   = (const float*)d_in[6];

    char* ws = (char*)d_ws;
    __hip_bfloat16* xbf  = (__hip_bfloat16*)ws;                          // 32 MB
    __hip_bfloat16* wqkt = (__hip_bfloat16*)(ws + (size_t)33554432);     //  4 MB
    __hip_bfloat16* wvt  = (__hip_bfloat16*)(ws + (size_t)37748736);     //  2 MB

    float* out_q = (float*)d_out;
    float* out_k = out_q + (size_t)BATCH * HEADS * NE * 64;
    float* out_v = out_k + (size_t)BATCH * HEADS * NE * 64;

    ln_bf16_kernel<<<ROWS, 256, 0, stream>>>(inp, gamma, beta, xbf);
    wt_bf16_kernel<<<dim3(2 * DIM / 32, DIM / 32), 256, 0, stream>>>(W_qk, wqkt, DIM, 2 * DIM);
    wt_bf16_kernel<<<dim3(DIM / 32, DIM / 32),     256, 0, stream>>>(W_v,  wvt,  DIM, DIM);

    gemm_qkv_kernel<<<dim3(2 * DIM / TN, ROWS / TM), 256, 0, stream>>>(
        xbf, wqkt, b_qk, out_q, out_k, out_v, /*mode=*/0);
    gemm_qkv_kernel<<<dim3(DIM / TN, ROWS / TM), 256, 0, stream>>>(
        xbf, wvt, b_v, out_q, out_k, out_v, /*mode=*/1);
}
